// Denoising_GNN_55800215110128
// MI455X (gfx1250) — matrix operand order repaired and checked by reading, not compiled, not run on hardware
//
#include <hip/hip_runtime.h>

// ---------------------------------------------------------------------------
// Types for WMMA (CDNA5 / gfx1250, wave32)
// ---------------------------------------------------------------------------
typedef __attribute__((ext_vector_type(16))) __bf16          v16bf;
typedef __attribute__((ext_vector_type(8)))  float           v8f;
typedef __attribute__((ext_vector_type(16))) unsigned short  v16u;
typedef __attribute__((ext_vector_type(8)))  unsigned short  v8u;
typedef __attribute__((ext_vector_type(4)))  int             v4i;

// Pointer types for the async global->LDS builtin:
//   __builtin_amdgcn_global_load_async_to_lds_b128(v4i global*, v4i local*, imm off, imm cpol)
typedef __attribute__((address_space(1))) v4i* gv4i_p;   // global (AS1)
typedef __attribute__((address_space(3))) v4i* lv4i_p;   // LDS (AS3)

// Async global->LDS copy path (gfx1250). Guarded: host pass / older toolchains
// fall back to synchronous b128 copies.
#if defined(__gfx1250__) && __has_builtin(__builtin_amdgcn_global_load_async_to_lds_b128) && __has_builtin(__builtin_amdgcn_s_wait_asynccnt)
#define USE_ASYNC_LDS 1
#else
#define USE_ASYNC_LDS 0
#endif

__device__ inline unsigned short f2bf(float f) {
  unsigned int u = __float_as_uint(f);
  u += 0x7FFFu + ((u >> 16) & 1u);      // round-to-nearest-even
  return (unsigned short)(u >> 16);
}

// ---------------------------------------------------------------------------
// Weight pre-convert: Wt[n*K + k] = bf16( W[k*Nc + n] )   (transposed, bf16)
// Weights are tiny (<= 512x512): negligible cost, makes GEMM staging b128.
// ---------------------------------------------------------------------------
__global__ __launch_bounds__(256) void wconv_kernel(
    const float* __restrict__ W, unsigned short* __restrict__ Wt, int K, int Nc)
{
  int i = blockIdx.x * 256 + threadIdx.x;
  if (i >= K * Nc) return;
  int n = i / K, k = i - n * K;
  Wt[i] = f2bf(W[(size_t)k * Nc + n]);
}

// ---------------------------------------------------------------------------
// Tiled WMMA GEMM: out[M,Nc] = act( A[M,K] @ W[K,Nc] + bias (+ residual) )
// Block tile 64x128, K-step 32, 256 threads = 8 waves.
// Wave layout: 4 row-bands x 2 col-halves; each wave: 16 rows x 64 cols
// = 4 WMMA tiles sharing one A fragment (A-reuse x4).
// B tile staged via async global->LDS (ASYNCcnt) when available.
// ---------------------------------------------------------------------------
#define TM 64
#define TN 128
#define TK 32

__global__ __launch_bounds__(256) void gemm_bf16_kernel(
    const float* __restrict__ A, const unsigned short* __restrict__ Wt, // [Nc][K] bf16
    const float* __restrict__ bias, const float* __restrict__ residual,
    float* __restrict__ out, int K, int Nc, int act)
{
  __shared__ __attribute__((aligned(16))) unsigned short lA[TM * TK]; // [m][k] bf16
  __shared__ __attribute__((aligned(16))) unsigned short lB[TN * TK]; // [n][k] bf16

  const int tid  = threadIdx.x;
  const int wave = tid >> 5;
  const int lane = tid & 31;
  const int row0 = blockIdx.x * TM;
  const int col0 = blockIdx.y * TN;
  const int trow = (wave & 3) * 16;        // wave's 16-row band
  const int tcb  = (wave >> 2) * 64;       // wave's 64-col band (4 tiles)

  v8f acc[4] = {{}, {}, {}, {}};

  // A fragment (16-bit A 16x32): lane<16 -> K {0..7,16..23}; lane>=16 -> {8..15,24..31}
  const int mrow = trow + (lane & 15);
  const int akb  = (lane < 16) ? 0 : 8;
  // B fragment (16-bit B 32x16): col = lane&15; lanes<16 -> K {0..7,16..23}; lanes>=16 -> {8..15,24..31}
  const int ncol = lane & 15;
  const int bkb  = (lane < 16) ? 0 : 8;

  // staging assignments
  const int ar = tid >> 2;                 // 0..63  (A row)
  const int ac = (tid & 3) * 8;            // 0,8,16,24 (A col group of 8)
  const int bn = tid >> 1;                 // 0..127 (B row = output col)
  const int bk = (tid & 1) * 16;           // 0,16  (B col group of 16)

  const float*          pa = &A [(size_t)(row0 + ar) * K + ac];
  const unsigned short* pb = &Wt[(size_t)(col0 + bn) * K + bk];
  unsigned short*       lb = &lB[bn * TK + bk];

  for (int k0 = 0; k0 < K; k0 += TK) {
    // ---- stage B: raw bf16 copy global -> LDS ----
#if USE_ASYNC_LDS
    __builtin_amdgcn_global_load_async_to_lds_b128(
        (gv4i_p)(pb + k0), (lv4i_p)lb, 0, 0);
    __builtin_amdgcn_global_load_async_to_lds_b128(
        (gv4i_p)(pb + k0), (lv4i_p)lb, 16, 0);   // +16B on both sides
#else
    v8u w0 = *(const v8u*)(pb + k0);
    v8u w1 = *(const v8u*)(pb + k0 + 8);
    *(v8u*)(lb)     = w0;
    *(v8u*)(lb + 8) = w1;
#endif

    // ---- stage A: 8 f32 -> 8 bf16, one ds_store_b128 per thread ----
    float4 f0 = *(const float4*)(pa + k0);
    float4 f1 = *(const float4*)(pa + k0 + 4);
    v8u av;
    av[0] = f2bf(f0.x); av[1] = f2bf(f0.y); av[2] = f2bf(f0.z); av[3] = f2bf(f0.w);
    av[4] = f2bf(f1.x); av[5] = f2bf(f1.y); av[6] = f2bf(f1.z); av[7] = f2bf(f1.w);
    *(v8u*)&lA[ar * TK + ac] = av;

    // prefetch next K-slice (global_prefetch_b8)
    if (k0 + TK < K) {
      __builtin_prefetch(pa + k0 + TK, 0, 1);
      __builtin_prefetch(pb + k0 + TK, 0, 1);
    }

#if USE_ASYNC_LDS
    __builtin_amdgcn_s_wait_asynccnt(0);
#endif
    __syncthreads();

    v8u alo = *(const v8u*)&lA[mrow * TK + akb];
    v8u ahi = *(const v8u*)&lA[mrow * TK + 16 + akb];
    v16bf afr = __builtin_bit_cast(v16bf,
        __builtin_shufflevector(alo, ahi, 0,1,2,3,4,5,6,7,8,9,10,11,12,13,14,15));

#pragma unroll
    for (int t = 0; t < 4; ++t) {
      int nb = (tcb + t * 16 + ncol) * TK;
      v8u bl = *(const v8u*)&lB[nb + bkb];
      v8u bh = *(const v8u*)&lB[nb + bkb + 16];
      v16bf bfr = __builtin_bit_cast(v16bf,
          __builtin_shufflevector(bl, bh, 0,1,2,3,4,5,6,7,8,9,10,11,12,13,14,15));
      acc[t] = __builtin_amdgcn_wmma_f32_16x16x32_bf16(
                   false, afr, false, bfr, (short)0, acc[t], false, false);
    }
    __syncthreads();
  }

  // ---- epilogue: C/D layout -> lane holds col (lane&15), rows trow+j (+8 upper) ----
#pragma unroll
  for (int t = 0; t < 4; ++t) {
    int c = col0 + tcb + t * 16 + ncol;
    float bv = bias[c];
#pragma unroll
    for (int j = 0; j < 8; ++j) {
      int m = trow + ((lane < 16) ? j : j + 8);
      size_t off = (size_t)(row0 + m) * Nc + c;
      float v = acc[t][j] + bv;
      if (residual) v += residual[off];
      if (act) v = fmaxf(v, 0.f);
      out[off] = v;
    }
  }
}

// ---------------------------------------------------------------------------
// LayerNorm + ReLU, in place, one block (256 threads) per row
// ---------------------------------------------------------------------------
__global__ __launch_bounds__(256) void ln_relu_kernel(
    float* __restrict__ r, const float* __restrict__ g,
    const float* __restrict__ b, int D)
{
  float* p = r + (size_t)blockIdx.x * D;
  float s = 0.f, ss = 0.f;
  for (int i = threadIdx.x; i < D; i += 256) {
    float v = p[i];
    s += v; ss += v * v;
  }
  for (int o = 16; o > 0; o >>= 1) {
    s  += __shfl_xor(s,  o, 32);
    ss += __shfl_xor(ss, o, 32);
  }
  __shared__ float sh1[8], sh2[8];
  int wv = threadIdx.x >> 5, ln = threadIdx.x & 31;
  if (ln == 0) { sh1[wv] = s; sh2[wv] = ss; }
  __syncthreads();
  if (threadIdx.x == 0) {
    float t1 = 0.f, t2 = 0.f;
    for (int i = 0; i < 8; ++i) { t1 += sh1[i]; t2 += sh2[i]; }
    sh1[0] = t1; sh2[0] = t2;
  }
  __syncthreads();
  float mean = sh1[0] / (float)D;
  float var  = sh2[0] / (float)D - mean * mean;
  float inv  = rsqrtf(var + 1e-5f);
  for (int i = threadIdx.x; i < D; i += 256) {
    float v = (p[i] - mean) * inv * g[i] + b[i];
    p[i] = fmaxf(v, 0.f);
  }
}

// ---------------------------------------------------------------------------
// Elementwise helpers
// ---------------------------------------------------------------------------
__global__ __launch_bounds__(256) void fill_kernel(float* p, float v, unsigned long long n) {
  unsigned long long i = (unsigned long long)blockIdx.x * 256ull + threadIdx.x;
  if (i < n) p[i] = v;
}

__global__ __launch_bounds__(256) void add_kernel(const float* __restrict__ a,
                                                  const float* __restrict__ b,
                                                  float* __restrict__ o,
                                                  unsigned long long n) {
  unsigned long long i = (unsigned long long)blockIdx.x * 256ull + threadIdx.x;
  if (i < n) o[i] = a[i] + b[i];
}

// ---------------------------------------------------------------------------
// GATv2 edge kernels
// ---------------------------------------------------------------------------
__device__ inline void atomicMaxF(float* addr, float v) {
  if (v >= 0.f) atomicMax((int*)addr, __float_as_int(v));
  else          atomicMin((unsigned int*)addr, __float_as_uint(v));
}

template <int D>
__global__ __launch_bounds__(256) void gat_logits_kernel(
    const float* __restrict__ fs, const float* __restrict__ fd,
    const float* __restrict__ attn, const int* __restrict__ src,
    const int* __restrict__ dst, float* __restrict__ logits,
    float* __restrict__ mmax, int E)
{
  constexpr int H = 256 / D;
  int idx = blockIdx.x * 256 + threadIdx.x;
  if (idx >= E * H) return;
  int e = idx / H, h = idx - e * H;
  int s = src[e], d = dst[e];
  const float* ps = fs + (size_t)s * 256 + h * D;
  const float* pd = fd + (size_t)d * 256 + h * D;
  const float* pa = attn + h * D;
  float acc = 0.f;
#pragma unroll
  for (int i = 0; i < D; i += 4) {
    float4 a4 = *(const float4*)(ps + i);
    float4 b4 = *(const float4*)(pd + i);
    float4 w4 = *(const float4*)(pa + i);
    float v;
    v = a4.x + b4.x; v = (v > 0.f) ? v : 0.2f * v; acc += v * w4.x;
    v = a4.y + b4.y; v = (v > 0.f) ? v : 0.2f * v; acc += v * w4.y;
    v = a4.z + b4.z; v = (v > 0.f) ? v : 0.2f * v; acc += v * w4.z;
    v = a4.w + b4.w; v = (v > 0.f) ? v : 0.2f * v; acc += v * w4.w;
  }
  logits[idx] = acc;
  atomicMaxF(&mmax[(size_t)d * H + h], acc);
}

__global__ __launch_bounds__(256) void gat_exp_kernel(
    float* __restrict__ logits /* in: logits, out: exp */,
    const int* __restrict__ dst, const float* __restrict__ mmax,
    float* __restrict__ den, int E, int H)
{
  int idx = blockIdx.x * 256 + threadIdx.x;
  if (idx >= E * H) return;
  int e = idx / H, h = idx - e * H;
  int d = dst[e];
  float v = __expf(logits[idx] - mmax[(size_t)d * H + h]);
  logits[idx] = v;
  atomicAdd(&den[(size_t)d * H + h], v);
}

__global__ __launch_bounds__(256) void gat_scatter_kernel(
    const float* __restrict__ fs, const int* __restrict__ src,
    const int* __restrict__ dst, const float* __restrict__ ex,
    const float* __restrict__ den, float* __restrict__ out, int H, int D)
{
  int e = blockIdx.x;
  int f = threadIdx.x;                  // feature 0..255 (H*D == 256 always)
  int h = f / D;
  int s = src[e], d = dst[e];
  float alpha = ex[(size_t)e * H + h] / den[(size_t)d * H + h];
  atomicAdd(&out[(size_t)d * 256 + f], fs[(size_t)s * 256 + f] * alpha);
}

// ---------------------------------------------------------------------------
// Host orchestration
// ---------------------------------------------------------------------------
extern "C" void kernel_launch(void* const* d_in, const int* in_sizes, int n_in,
                              void* d_out, int out_size, void* d_ws, size_t ws_size,
                              hipStream_t stream) {
  (void)in_sizes; (void)n_in; (void)out_size; (void)ws_size;
  const int N = 40000, E = 400000;

  const float* x_t = (const float*)d_in[0];
  const int*   src = (const int*)d_in[2];
  const int*   dst = (const int*)d_in[3];

  auto P = [&](int i) { return (const float*)d_in[i]; };

  struct Gat { const float *attn, *db, *dw, *sb, *sw; };
  auto gat_at = [&](int b) { Gat g; g.attn = P(b); g.db = P(b+1); g.dw = P(b+2);
                             g.sb = P(b+3); g.sw = P(b+4); return g; };
  struct Mlp { const float *h1b,*h1w,*h2b,*h2w,*inb,*inw,*lnb,*lng,*outb,*outw; };
  auto mlp_at = [&](int b) { Mlp m; m.h1b=P(b); m.h1w=P(b+1); m.h2b=P(b+2); m.h2w=P(b+3);
                             m.inb=P(b+4); m.inw=P(b+5); m.lnb=P(b+6); m.lng=P(b+7);
                             m.outb=P(b+8); m.outw=P(b+9); return m; };

  // JAX pytree order: sorted dict keys -> down, mlp_in, mlp_mid, mlp_out, up
  Gat down[3] = { gat_at(4),  gat_at(9),  gat_at(14) };
  Mlp mIn  = mlp_at(19);
  Mlp mMid = mlp_at(29);
  Mlp mOut = mlp_at(39);
  Gat up[3]   = { gat_at(49), gat_at(54), gat_at(59) };

  // Workspace bump allocator
  float* wsp = (float*)d_ws;
  auto alloc = [&](size_t n) { float* p = wsp; wsp += n; return p; };
  float* bufA = alloc((size_t)N * 512);   // mlp "h"
  float* bufB = alloc((size_t)N * 512);   // mlp "r"
  float* hcur = alloc((size_t)N * 256);
  float* hid0 = alloc((size_t)N * 256);
  float* hid1 = alloc((size_t)N * 256);
  float* hid2 = alloc((size_t)N * 256);
  float* fsb  = alloc((size_t)N * 256);
  float* fdb  = alloc((size_t)N * 256);
  float* tsum = alloc((size_t)N * 256);
  float* gbuf = alloc((size_t)N * 256);
  float* elog = alloc((size_t)E * 8);
  float* mbuf = alloc((size_t)N * 8);
  float* dden = alloc((size_t)N * 8);
  unsigned short* wtmp = (unsigned short*)alloc((size_t)512 * 512 / 2); // bf16 W^T scratch

  auto gemm = [&](const float* A, const float* W, const float* bias,
                  const float* res, float* out, int K, int Nc, int act) {
    int tot = K * Nc;
    wconv_kernel<<<(tot + 255) / 256, 256, 0, stream>>>(W, wtmp, K, Nc);
    dim3 grid(N / TM, Nc / TN);
    gemm_bf16_kernel<<<grid, 256, 0, stream>>>(A, wtmp, bias, res, out, K, Nc, act);
  };
  auto fill = [&](float* p, size_t n, float v) {
    fill_kernel<<<(unsigned)((n + 255) / 256), 256, 0, stream>>>(p, v, n);
  };

  auto run_mlp = [&](const Mlp& m, const float* x, int din, int dh, int dout, float* out) {
    gemm(x,    m.inw, m.inb, nullptr, bufA, din, dh, 0);       // h = x@Win + b
    gemm(bufA, m.h1w, m.h1b, nullptr, bufB, dh,  dh, 0);       // r = h@W1 + b
    ln_relu_kernel<<<N, 256, 0, stream>>>(bufB, m.lng, m.lnb, dh);
    gemm(bufB, m.h2w, m.h2b, bufA, bufA, dh, dh, 0);           // h = h + r@W2 + b (in-place resid)
    gemm(bufA, m.outw, m.outb, nullptr, out, dh, dout, 1);     // relu(h@Wout + b)
  };

  const float NEG_INF = -__builtin_huge_valf();
  auto run_gat = [&](const Gat& g, const float* x, float* out, int H, int D) {
    gemm(x, g.sw, g.sb, nullptr, fsb, 256, 256, 0);            // fs
    gemm(x, g.dw, g.db, nullptr, fdb, 256, 256, 0);            // fd
    fill(mbuf, (size_t)N * H, NEG_INF);
    fill(dden, (size_t)N * H, 0.f);
    int tot = E * H;
    int blk = (tot + 255) / 256;
    if (D == 32)
      gat_logits_kernel<32><<<blk, 256, 0, stream>>>(fsb, fdb, g.attn, src, dst, elog, mbuf, E);
    else
      gat_logits_kernel<256><<<blk, 256, 0, stream>>>(fsb, fdb, g.attn, src, dst, elog, mbuf, E);
    gat_exp_kernel<<<blk, 256, 0, stream>>>(elog, dst, mbuf, dden, E, H);
    fill(out, (size_t)N * 256, 0.f);
    gat_scatter_kernel<<<E, 256, 0, stream>>>(fsb, src, dst, elog, dden, out, H, D);
  };

  // ---------------- forward ----------------
  run_mlp(mIn, x_t, 256, 512, 256, hcur);

  run_gat(down[0], hcur, hid0, 8, 32);
  run_gat(down[1], hid0, hid1, 8, 32);
  run_gat(down[2], hid1, hid2, 8, 32);

  run_mlp(mMid, hid2, 256, 256, 256, hcur);

  unsigned long long nf = (unsigned long long)N * 256ull;
  unsigned nblk = (unsigned)((nf + 255) / 256);

  add_kernel<<<nblk, 256, 0, stream>>>(hcur, hid2, tsum, nf);
  run_gat(up[0], tsum, gbuf, 8, 32);
  add_kernel<<<nblk, 256, 0, stream>>>(gbuf, hid1, tsum, nf);
  run_gat(up[1], tsum, gbuf, 8, 32);
  add_kernel<<<nblk, 256, 0, stream>>>(gbuf, hid0, tsum, nf);

  float* h_final = (float*)d_out + (size_t)N * 256;            // second output: h
  run_gat(up[2], tsum, h_final, 1, 256);

  run_mlp(mOut, h_final, 256, 256, 256, (float*)d_out);        // first output: out
}
